// GraphLinkGAT_13013750906975
// MI455X (gfx1250) — compile-verified, matrix operand order checked by reading, not run on hardware
//
#include <hip/hip_runtime.h>
#include <stdint.h>

// ---------------------------------------------------------------------------
// Types for CDNA5 WMMA (wave32): A,B = 16 bf16 per lane; C/D = 8 f32 per lane
// ---------------------------------------------------------------------------
typedef __attribute__((ext_vector_type(16))) __bf16 v16bf;
typedef __attribute__((ext_vector_type(8)))  float  v8f;

__device__ __forceinline__ unsigned short f2bf(float f) {
    unsigned u = __float_as_uint(f);
    u += 0x7fffu + ((u >> 16) & 1u);          // round-to-nearest-even
    return (unsigned short)(u >> 16);
}
// monotone float <-> uint encoding for atomicMax on signed floats
__device__ __forceinline__ unsigned fenc(float f) {
    unsigned b = __float_as_uint(f);
    return (b & 0x80000000u) ? ~b : (b | 0x80000000u);
}
__device__ __forceinline__ float fdec(unsigned k) {
    unsigned b = (k & 0x80000000u) ? (k & 0x7fffffffu) : ~k;
    return __uint_as_float(b);
}

// CDNA5 async copy: global -> LDS, tracked by ASYNCcnt (CDNA5 ISA)
__device__ __forceinline__ void async_ld_b128(unsigned lds_off, const void* gsrc) {
    asm volatile("global_load_async_to_lds_b128 %0, %1, off"
                 :: "v"(lds_off), "v"((uint64_t)(uintptr_t)gsrc) : "memory");
}
__device__ __forceinline__ void wait_async0() {
    asm volatile("s_wait_asynccnt 0" ::: "memory");
}

// ---------------------------------------------------------------------------
// Convert fp32 weight [K x Nc] -> bf16 transposed [Nc x K] (row = out column)
// ---------------------------------------------------------------------------
__global__ void k_conv_t(const float* __restrict__ W, unsigned short* __restrict__ Wt,
                         int K, int Nc) {
    int i = blockIdx.x * blockDim.x + threadIdx.x;
    if (i >= K * Nc) return;
    int k = i / Nc, n = i - k * Nc;
    Wt[(size_t)n * K + k] = f2bf(W[i]);
}

// ---------------------------------------------------------------------------
// Time embedding (swish) + concat with x -> bf16 activation [N x 384]
// ---------------------------------------------------------------------------
__global__ void k_te_concat(const float* __restrict__ x, const float* __restrict__ t,
                            const float* __restrict__ tew, const float* __restrict__ teb,
                            unsigned short* __restrict__ Abf, int N) {
    int i = blockIdx.x * blockDim.x + threadIdx.x;
    if (i >= N * 384) return;
    int node = i / 384, j = i - node * 384;
    float v;
    if (j < 128) {
        v = x[(size_t)node * 128 + j];
    } else {
        int jj = j - 128;
        float a  = t[node] * (1.0f / 1000.0f);
        float ph = a * 1.57079632679f;
        float u  = sinf(ph) * tew[jj] + cosf(ph) * tew[256 + jj] + a * tew[512 + jj] + teb[jj];
        v = u / (1.0f + __expf(-u));   // swish
    }
    Abf[i] = f2bf(v);
}

// ---------------------------------------------------------------------------
// LDS-staged, async-copy, double-buffered WMMA GEMM for Nc == 256.
// Block = 256 threads (8 waves) -> 32 rows x 256 cols.
// Wave w: row subtile (w>>2), col group 64*(w&3); 4 accumulators of 16x16.
// A row-major bf16 [MxK]; Bt row-major bf16 [256xK] holding B^T. K%32==0.
// ---------------------------------------------------------------------------
__global__ __launch_bounds__(256)
void k_gemm_lds(const unsigned short* __restrict__ A, const unsigned short* __restrict__ Bt,
                const float* __restrict__ bias, float* __restrict__ C,
                unsigned short* __restrict__ Cbf, int M, int K, int relu) {
    __shared__ unsigned short sA[2][32 * 32];    // 2 x 2KB
    __shared__ unsigned short sB[2][256 * 32];   // 2 x 16KB

    const int tid  = threadIdx.x;
    const int lane = tid & 31;
    const int w    = tid >> 5;          // wave 0..7
    const int rsub = w >> 2;            // 0..1 : 16-row subtile
    const int cg   = w & 3;             // 0..3 : 64-col group
    const int mr   = lane & 15;
    const int koff = (lane >> 4) << 3;  // lanes 16..31 take K+8 half

    const int rowBase  = blockIdx.x * 32;
    const bool rowOK   = (rowBase + rsub * 16) < M;   // tail guard (M%16==0)

    union { v8f v; float f[8]; } acc[4];
#pragma unroll
    for (int a = 0; a < 4; ++a)
#pragma unroll
        for (int r = 0; r < 8; ++r) acc[a].f[r] = 0.0f;

    // ---- stage issuer (async global->LDS) ----
    auto stage = [&](int buf, int kk) {
        // B slab: 256 cols x 32 k ; one col per thread, 4 x b128
        {
            const unsigned short* src = Bt + (size_t)tid * K + kk;
            unsigned dst = (unsigned)(uintptr_t)&sB[buf][tid * 32];
#pragma unroll
            for (int c = 0; c < 4; ++c)
                async_ld_b128(dst + c * 16, src + c * 8);
        }
        // A slab: 32 rows x 32 k ; threads 0..127, row=tid>>2, chunk=tid&3
        if (tid < 128) {
            int row = tid >> 2, c = tid & 3;
            if (rowBase + row < M) {
                const unsigned short* src = A + (size_t)(rowBase + row) * K + kk + c * 8;
                unsigned dst = (unsigned)(uintptr_t)&sA[buf][row * 32 + c * 8];
                async_ld_b128(dst, src);
            }
        }
    };

    int buf = 0;
    stage(0, 0);
    for (int kk = 0; kk < K; kk += 32) {
        wait_async0();            // our slab's async copies done
        __syncthreads();          // visible to all waves
        if (kk + 32 < K) stage(buf ^ 1, kk + 32);

        if (rowOK) {
            union { v16bf v; unsigned u[8]; } fa;
            const unsigned* Ar = (const unsigned*)&sA[buf][(rsub * 16 + mr) * 32];
#pragma unroll
            for (int vv = 0; vv < 8; ++vv) {
                const int k0 = ((vv >> 2) << 4) + ((vv & 3) << 1) + koff;
                fa.u[vv] = Ar[k0 >> 1];
            }
#pragma unroll
            for (int a = 0; a < 4; ++a) {
                union { v16bf v; unsigned u[8]; } fb;
                const unsigned* Br = (const unsigned*)&sB[buf][(cg * 64 + a * 16 + mr) * 32];
#pragma unroll
                for (int vv = 0; vv < 8; ++vv) {
                    const int k0 = ((vv >> 2) << 4) + ((vv & 3) << 1) + koff;
                    fb.u[vv] = Br[k0 >> 1];
                }
                acc[a].v = __builtin_amdgcn_wmma_f32_16x16x32_bf16(false, fa.v, false, fb.v,
                                                                   (short)0, acc[a].v,
                                                                   false, false);
            }
        }
        __syncthreads();          // everyone done reading buf before it is refilled
        buf ^= 1;
    }

    if (!rowOK) return;
    const int rbase = rowBase + rsub * 16 + ((lane >> 4) << 3);
#pragma unroll
    for (int a = 0; a < 4; ++a) {
        const int col  = cg * 64 + a * 16 + (lane & 15);
        const float bv = bias ? bias[col] : 0.0f;
#pragma unroll
        for (int r = 0; r < 8; ++r) {
            float v = acc[a].f[r] + bv;
            if (relu) v = fmaxf(v, 0.0f);
            const size_t idx = (size_t)(rbase + r) * 256 + col;
            if (C)   C[idx]   = v;
            if (Cbf) Cbf[idx] = f2bf(v);
        }
    }
}

// ---------------------------------------------------------------------------
// Direct-load WMMA GEMM (used for small Nc, e.g. 32). One wave = 16x16 tile.
// ---------------------------------------------------------------------------
__global__ __launch_bounds__(256)
void k_gemm(const unsigned short* __restrict__ A, const unsigned short* __restrict__ Bt,
            const float* __restrict__ bias, float* __restrict__ C,
            unsigned short* __restrict__ Cbf, int M, int Nc, int K, int relu) {
    const int lane = threadIdx.x & 31;
    const int wid  = (blockIdx.x * blockDim.x + threadIdx.x) >> 5;
    const int ntN  = Nc >> 4;
    const int tM   = wid / ntN;
    const int tN   = wid - tM * ntN;
    if (tM * 16 >= M) return;

    const int mr   = lane & 15;
    const int koff = (lane >> 4) << 3;

    const unsigned* __restrict__ Arow = (const unsigned*)(A + (size_t)(tM * 16 + mr) * K);
    const unsigned* __restrict__ Brow = (const unsigned*)(Bt + (size_t)(tN * 16 + mr) * K);

    union { v16bf v; unsigned u[8]; } fa, fb;
    union { v8f   v; float    f[8]; } acc;
#pragma unroll
    for (int r = 0; r < 8; ++r) acc.f[r] = 0.0f;

    for (int kk = 0; kk < K; kk += 32) {
#pragma unroll
        for (int vv = 0; vv < 8; ++vv) {
            const int k0 = kk + ((vv >> 2) << 4) + ((vv & 3) << 1) + koff;
            fa.u[vv] = Arow[k0 >> 1];
            fb.u[vv] = Brow[k0 >> 1];
        }
        acc.v = __builtin_amdgcn_wmma_f32_16x16x32_bf16(false, fa.v, false, fb.v,
                                                        (short)0, acc.v, false, false);
    }

    const int col   = tN * 16 + (lane & 15);
    const float bv  = bias ? bias[col] : 0.0f;
    const int rbase = tM * 16 + ((lane >> 4) << 3);
#pragma unroll
    for (int r = 0; r < 8; ++r) {
        float v = acc.f[r] + bv;
        if (relu) v = fmaxf(v, 0.0f);
        const size_t idx = (size_t)(rbase + r) * Nc + col;
        if (C)   C[idx]   = v;
        if (Cbf) Cbf[idx] = f2bf(v);
    }
}

// ---------------------------------------------------------------------------
// Edge pass 1: e[e,h] = sum_d leaky(hl[s]+hr[d]) * att ; segment-max into Mb
// ---------------------------------------------------------------------------
__global__ void k_edge_e_max(const int* __restrict__ ei, const float* __restrict__ HL,
                             const float* __restrict__ HR, const float* __restrict__ att,
                             float* __restrict__ Eb, unsigned* __restrict__ Mb,
                             int E, int Etot) {
    const int lane = threadIdx.x & 31;
    const int e    = (blockIdx.x * blockDim.x + threadIdx.x) >> 5;
    if (e >= Etot) return;
    int s, d;
    if (e < E) { s = ei[e]; d = ei[E + e]; } else { s = d = e - E; }
    const float* hl = HL + (size_t)s * 256;
    const float* hr = HR + (size_t)d * 256;
#pragma unroll
    for (int h = 0; h < 8; ++h) {
        float xv = hl[h * 32 + lane] + hr[h * 32 + lane];
        xv = (xv > 0.0f) ? xv : 0.2f * xv;            // leaky relu
        float v = xv * att[h * 32 + lane];
        v += __shfl_xor(v, 16); v += __shfl_xor(v, 8); v += __shfl_xor(v, 4);
        v += __shfl_xor(v, 2);  v += __shfl_xor(v, 1);
        if (lane == 0) {
            Eb[(size_t)e * 8 + h] = v;
            atomicMax(&Mb[(size_t)d * 8 + h], fenc(v));
        }
    }
}

// Edge pass 2: e' = exp(e - max); segment-sum into Zb. One thread per (edge,head)
__global__ void k_edge_softmax(const int* __restrict__ ei, float* __restrict__ Eb,
                               const unsigned* __restrict__ Mb, float* __restrict__ Zb,
                               int E, int Etot) {
    int i = blockIdx.x * blockDim.x + threadIdx.x;
    if (i >= Etot * 8) return;
    int e = i >> 3, h = i & 7;
    int d = (e < E) ? ei[E + e] : (e - E);
    float m  = fdec(Mb[(size_t)d * 8 + h]);
    float ex = __expf(Eb[i] - m);
    Eb[i] = ex;
    atomicAdd(&Zb[(size_t)d * 8 + h], ex);
}

// Edge pass 3: G[dst] += alpha * hl[src]. One wave per edge.
__global__ void k_edge_msg(const int* __restrict__ ei, const float* __restrict__ Eb,
                           const float* __restrict__ Zb, const float* __restrict__ HL,
                           float* __restrict__ G, int E, int Etot) {
    const int lane = threadIdx.x & 31;
    const int e    = (blockIdx.x * blockDim.x + threadIdx.x) >> 5;
    if (e >= Etot) return;
    int s, d;
    if (e < E) { s = ei[e]; d = ei[E + e]; } else { s = d = e - E; }
#pragma unroll
    for (int h = 0; h < 8; ++h) {
        float alpha = Eb[(size_t)e * 8 + h] / (Zb[(size_t)d * 8 + h] + 1e-16f);
        atomicAdd(&G[(size_t)d * 256 + h * 32 + lane],
                  alpha * HL[(size_t)s * 256 + h * 32 + lane]);
    }
}

// S = Aa + Bb (+bias[col]); block-reduce sum & sumsq into red[0..1] (doubles)
__global__ void k_sum_reduce(const float* __restrict__ Aa, const float* __restrict__ Bb,
                             const float* __restrict__ bias, float* __restrict__ S,
                             double* __restrict__ red, int total) {
    __shared__ float s1[256], s2[256];
    int i = blockIdx.x * blockDim.x + threadIdx.x;
    float v = 0.0f;
    if (i < total) {
        v = Aa[i] + Bb[i] + (bias ? bias[i & 255] : 0.0f);
        S[i] = v;
    }
    s1[threadIdx.x] = v; s2[threadIdx.x] = v * v;
    __syncthreads();
    for (int st = 128; st > 0; st >>= 1) {
        if ((int)threadIdx.x < st) {
            s1[threadIdx.x] += s1[threadIdx.x + st];
            s2[threadIdx.x] += s2[threadIdx.x + st];
        }
        __syncthreads();
    }
    if (threadIdx.x == 0) {
        atomicAdd(&red[0], (double)s1[0]);
        atomicAdd(&red[1], (double)s2[0]);
    }
}

// graph layernorm apply; outputs f32 (optional) + bf16
__global__ void k_ln_apply(const float* __restrict__ S, const float* __restrict__ w,
                           const float* __restrict__ b, const double* __restrict__ red,
                           float* __restrict__ Of, unsigned short* __restrict__ Obf,
                           int total) {
    int i = blockIdx.x * blockDim.x + threadIdx.x;
    if (i >= total) return;
    double inv = 1.0 / (double)total;
    double mu  = red[0] * inv;
    double var = red[1] * inv - mu * mu;
    float rs   = rsqrtf((float)var + 1e-5f);
    float v    = ((float)((double)S[i] - mu)) * rs * w[i & 255] + b[i & 255];
    if (Of) Of[i] = v;
    Obf[i] = f2bf(v);
}

// link head: sigmoid( (h[a]*h[b]) . link_w + link_b )
__global__ void k_link(const int* __restrict__ li, const float* __restrict__ Hn,
                       const float* __restrict__ lw, const float* __restrict__ lb,
                       float* __restrict__ out, int EL) {
    int i = blockIdx.x * blockDim.x + threadIdx.x;
    if (i >= EL) return;
    int a = li[i], b = li[EL + i];
    const float4* pa = (const float4*)(Hn + (size_t)a * 32);
    const float4* pb = (const float4*)(Hn + (size_t)b * 32);
    const float4* pw = (const float4*)lw;
    float acc = 0.0f;
#pragma unroll
    for (int q = 0; q < 8; ++q) {
        float4 va = pa[q], vb = pb[q], vw = pw[q];
        acc += va.x * vb.x * vw.x + va.y * vb.y * vw.y
             + va.z * vb.z * vw.z + va.w * vb.w * vw.w;
    }
    acc += lb[0];
    out[i] = 1.0f / (1.0f + __expf(-acc));
}

// ---------------------------------------------------------------------------
extern "C" void kernel_launch(void* const* d_in, const int* in_sizes, int n_in,
                              void* d_out, int out_size, void* d_ws, size_t ws_size,
                              hipStream_t stream) {
    const float* x        = (const float*)d_in[0];
    const float* t        = (const float*)d_in[1];
    const int*   ei       = (const int*)  d_in[2];
    const int*   li       = (const int*)  d_in[3];
    const float* te_w     = (const float*)d_in[4];
    const float* te_b     = (const float*)d_in[5];
    const float* pre1_w   = (const float*)d_in[6];
    const float* pre1_b   = (const float*)d_in[7];
    const float* pre2_w   = (const float*)d_in[8];
    const float* pre2_b   = (const float*)d_in[9];
    const float* lin_l    = (const float*)d_in[10];
    const float* lin_l_b  = (const float*)d_in[11];
    const float* lin_r    = (const float*)d_in[12];
    const float* lin_r_b  = (const float*)d_in[13];
    const float* att      = (const float*)d_in[14];
    const float* gat_bias = (const float*)d_in[15];
    const float* n1w      = (const float*)d_in[16];
    const float* n1b      = (const float*)d_in[17];
    const float* d2w      = (const float*)d_in[18];
    const float* d2b      = (const float*)d_in[19];
    const float* d3w      = (const float*)d_in[20];
    const float* d3b      = (const float*)d_in[21];
    const float* n3w      = (const float*)d_in[22];
    const float* n3b      = (const float*)d_in[23];
    const float* p1w      = (const float*)d_in[24];
    const float* p1b      = (const float*)d_in[25];
    const float* p2w      = (const float*)d_in[26];
    const float* p2b      = (const float*)d_in[27];
    const float* lw       = (const float*)d_in[28];
    const float* lb       = (const float*)d_in[29];
    float* out            = (float*)d_out;

    const int N    = in_sizes[1];
    const int E    = in_sizes[2] / 2;
    const int EL   = in_sizes[3] / 2;
    const int Etot = E + N;
    const int total = N * 256;

    // ---- carve workspace -------------------------------------------------
    char* p = (char*)d_ws;
    auto carve = [&](size_t bytes) -> char* {
        char* r = p; p += (bytes + 255) & ~(size_t)255; return r;
    };
    unsigned short* Wt  = (unsigned short*)carve((size_t)1286144 * 2); // bf16 weights
    unsigned short* Abf = (unsigned short*)carve((size_t)N * 384 * 2);
    unsigned short* Bbf = (unsigned short*)carve((size_t)N * 256 * 2);
    float* Hf = (float*)carve((size_t)N * 256 * 4);
    float* HL = (float*)carve((size_t)N * 256 * 4);
    float* HR = (float*)carve((size_t)N * 256 * 4);
    float* G  = (float*)carve((size_t)N * 256 * 4);
    float* Eb = (float*)carve((size_t)Etot * 8 * 4);
    unsigned* Mb = (unsigned*)carve((size_t)N * 8 * 4);
    float* Zb = (float*)carve((size_t)N * 8 * 4);
    double* red = (double*)carve(256);

    unsigned short* Wt_pre1  = Wt;                       // 384x256
    unsigned short* Wt_pre2  = Wt_pre1 + 98304;          // 256x256
    unsigned short* Wt_layer = Wt_pre2 + 65536;          // 16 x 256x256
    unsigned short* Wt_p1    = Wt_layer + 16 * 65536;    // 256x256
    unsigned short* Wt_p2    = Wt_p1 + 65536;            // 256x32

    auto conv = [&](const float* Wsrc, unsigned short* dst, int K, int Nc) {
        int n = K * Nc;
        k_conv_t<<<(n + 255) / 256, 256, 0, stream>>>(Wsrc, dst, K, Nc);
    };
    // Nc==256 path: LDS/async pipelined GEMM; otherwise direct-load GEMM
    auto gemm = [&](const unsigned short* A, const unsigned short* Bt, const float* bias,
                    float* C, unsigned short* Cbf, int M, int Nc, int K, int relu) {
        if (Nc == 256) {
            int blocks = (M / 16 + 1) / 2;       // 32 rows per block
            k_gemm_lds<<<blocks, 256, 0, stream>>>(A, Bt, bias, C, Cbf, M, K, relu);
        } else {
            int waves  = (M / 16) * (Nc / 16);
            int blocks = (waves * 32 + 255) / 256;
            k_gemm<<<blocks, 256, 0, stream>>>(A, Bt, bias, C, Cbf, M, Nc, K, relu);
        }
    };

    // ---- weight conversion ----------------------------------------------
    conv(pre1_w, Wt_pre1, 384, 256);
    conv(pre2_w, Wt_pre2, 256, 256);
    for (int i = 0; i < 4; ++i) {
        conv(lin_l + (size_t)i * 65536, Wt_layer + (size_t)(i * 4 + 0) * 65536, 256, 256);
        conv(lin_r + (size_t)i * 65536, Wt_layer + (size_t)(i * 4 + 1) * 65536, 256, 256);
        conv(d2w   + (size_t)i * 65536, Wt_layer + (size_t)(i * 4 + 2) * 65536, 256, 256);
        conv(d3w   + (size_t)i * 65536, Wt_layer + (size_t)(i * 4 + 3) * 65536, 256, 256);
    }
    conv(p1w, Wt_p1, 256, 256);
    conv(p2w, Wt_p2, 256, 32);

    // ---- time embedding + concat + pre-MLP ------------------------------
    k_te_concat<<<(N * 384 + 255) / 256, 256, 0, stream>>>(x, t, te_w, te_b, Abf, N);
    gemm(Abf, Wt_pre1, pre1_b, nullptr, Bbf, N, 256, 384, 0);     // h1 bf16 -> Bbf
    gemm(Bbf, Wt_pre2, pre2_b, Hf, Abf, N, 256, 256, 0);          // h f32 -> Hf, bf16 -> Abf

    const int edgeWaveBlocks = (Etot * 32 + 255) / 256;

    // ---- GAT layers ------------------------------------------------------
    for (int i = 0; i < 4; ++i) {
        hipMemsetAsync(Mb,  0, (size_t)N * 8 * 4, stream);   // fenc-min
        hipMemsetAsync(Zb,  0, (size_t)N * 8 * 4, stream);
        hipMemsetAsync(G,   0, (size_t)N * 256 * 4, stream);
        hipMemsetAsync(red, 0, 16, stream);

        gemm(Abf, Wt_layer + (size_t)(i * 4 + 0) * 65536, lin_l_b + i * 256, HL, nullptr, N, 256, 256, 0);
        gemm(Abf, Wt_layer + (size_t)(i * 4 + 1) * 65536, lin_r_b + i * 256, HR, nullptr, N, 256, 256, 0);

        k_edge_e_max<<<edgeWaveBlocks, 256, 0, stream>>>(ei, HL, HR, att + i * 256, Eb, Mb, E, Etot);
        k_edge_softmax<<<(Etot * 8 + 255) / 256, 256, 0, stream>>>(ei, Eb, Mb, Zb, E, Etot);
        k_edge_msg<<<edgeWaveBlocks, 256, 0, stream>>>(ei, Eb, Zb, HL, G, E, Etot);

        // h1 = LN(h + g + gat_bias)
        k_sum_reduce<<<(total + 255) / 256, 256, 0, stream>>>(Hf, G, gat_bias + i * 256, G, red, total);
        k_ln_apply<<<(total + 255) / 256, 256, 0, stream>>>(G, n1w + i * 256, n1b + i * 256, red,
                                                            nullptr, Abf, total);
        hipMemsetAsync(red, 0, 16, stream);

        // h2 = relu(h1@d2+b) ; h3 = h2@d3+b ; h = LN(h2+h3)
        gemm(Abf, Wt_layer + (size_t)(i * 4 + 2) * 65536, d2b + i * 256, HR, Bbf, N, 256, 256, 1);
        gemm(Bbf, Wt_layer + (size_t)(i * 4 + 3) * 65536, d3b + i * 256, HL, nullptr, N, 256, 256, 0);
        k_sum_reduce<<<(total + 255) / 256, 256, 0, stream>>>(HR, HL, nullptr, G, red, total);
        k_ln_apply<<<(total + 255) / 256, 256, 0, stream>>>(G, n3w + i * 256, n3b + i * 256, red,
                                                            Hf, Abf, total);
    }

    // ---- post MLP + link head -------------------------------------------
    gemm(Abf, Wt_p1, p1b, nullptr, Bbf, N, 256, 256, 0);
    gemm(Bbf, Wt_p2, p2b, HL, nullptr, N, 32, 256, 0);   // final h [N,32] in HL
    k_link<<<(EL + 255) / 256, 256, 0, stream>>>(li, HL, lw, lb, out, EL);
}
